// Embeddings_40398462386202
// MI455X (gfx1250) — matrix operand order repaired and checked by reading, not compiled, not run on hardware
//
#include <hip/hip_runtime.h>
#include <hip/hip_bf16.h>

// ---- problem constants (from reference) ----
#define S_    128
#define B_    32
#define W_    16             // chars per word
#define D_    300            // word emb dim
#define EE    64             // char emb dim (= conv groups)
#define FF    4              // channel multiplier
#define KK    3              // kernel size
#define OUTC  (EE*FF)        // 256 conv output channels
#define ROWF  (D_ + OUTC)    // 556 floats per output row
#define NT    (W_ - KK + 1)  // 14 valid conv positions
#define NPAIR (EE/2)         // 32 channel pairs per word-tile
#define NOUTP (2*NT*FF)      // 112 conv outputs per channel pair
#define NTILE (NOUTP/16)     // 7 WMMA N-tiles per pair
#define NWAVE 4              // waves per char block
#define BTAB_ELems (NPAIR*NTILE*32*16)        // f16 elements in B table
#define BTAB_BYTES (BTAB_ELems*2)             // 229376 B

typedef _Float16 v8h  __attribute__((ext_vector_type(8)));
typedef _Float16 v16h __attribute__((ext_vector_type(16)));
typedef float    v4f  __attribute__((ext_vector_type(4)));
typedef float    v8f  __attribute__((ext_vector_type(8)));

// ---------------------------------------------------------------------------
// Kernel 0: materialize the (word-tile independent) Toeplitz B fragments.
// Entry (pair p, tile tau, lane): 16 f16 = exactly the v16h that lane feeds
// to v_wmma as SRC1.  B[k = (lane<16 ? i : 16+i), n = lane&15]; n encodes
// (e_off, t, f); nonzero only at rows w = t..t+2 when the lane's K-half
// channel matches.  224 KB total -> L2-resident, shared by all 256 blocks.
// ---------------------------------------------------------------------------
__global__ __launch_bounds__(256) void build_btab_kernel(
    const float* __restrict__ cw, _Float16* __restrict__ btab) {
  const int idx = blockIdx.x * 256 + threadIdx.x;
  if (idx >= NPAIR * NTILE * 32) return;
  const int p    = idx / (NTILE * 32);
  const int rem  = idx % (NTILE * 32);
  const int tau  = rem / 32;
  const int lane = rem % 32;

  const int keoff = (lane < 16) ? 0 : 1;
  const int ng    = tau * 16 + (lane & 15);
  const int e_off = (ng >= NT * FF) ? 1 : 0;
  const int r2    = ng - e_off * (NT * FF);
  const int t     = r2 >> 2;
  const int f     = r2 & 3;
  const int ch    = 2 * p + e_off;

  float w0 = 0.f, w1 = 0.f, w2 = 0.f;
  if (e_off == keoff) {
    w0 = cw[0 * OUTC + ch * FF + f];
    w1 = cw[1 * OUTC + ch * FF + f];
    w2 = cw[2 * OUTC + ch * FF + f];
  }
  v16h bv;
#pragma unroll
  for (int i = 0; i < 16; ++i) {
    float v = 0.f;
    v = (i == t)     ? w0 : v;
    v = (i == t + 1) ? w1 : v;
    v = (i == t + 2) ? w2 : v;
    bv[i] = (_Float16)v;
  }
  *(v16h*)&btab[(size_t)idx * 16] = bv;
}

// ---------------------------------------------------------------------------
// Kernel 1: word embedding gather.  One block per (s,b) row; 75 float4
// copies (source row 1200 B and dest row 2224 B are both 16B aligned).
// This is the HBM-bound part of the op.
// ---------------------------------------------------------------------------
__global__ __launch_bounds__(96) void word_gather_kernel(
    const int* __restrict__ words, const float* __restrict__ table,
    float* __restrict__ out) {
  const int g   = blockIdx.x;            // g = s*B + b (matches [S,B,*] row)
  const int idx = words[g];              // uniform -> scalar load
  const v4f* __restrict__ src = (const v4f*)(table + (size_t)idx * D_);
  v4f* __restrict__ dst       = (v4f*)(out + (size_t)g * ROWF);
  const int t = threadIdx.x;
  if (t < D_ / 4) dst[t] = src[t];
}

// ---------------------------------------------------------------------------
// Kernel 2: char-CNN via v_wmma_f32_16x16x32_f16.
// 4 waves per block, one block per tile of 16 words; wave w owns pairs
// w*8 .. w*8+7 and a private ybuf slice.
//   A (16x32 f16): M = word, K = e_off*16 + w   (two channels per pair)
//   B (32x16 f16): precomputed Toeplitz fragment (global, L2-hot) or
//                  register-built fallback when d_ws is too small
//   D (16x16 f32): y[word,(e_off,t,f)] -> LDS -> max over t -> +bias -> out
// ---------------------------------------------------------------------------
__global__ __launch_bounds__(32 * NWAVE) void char_cnn_wmma_kernel(
    const int* __restrict__ chars, const float* __restrict__ ctab,
    const float* __restrict__ cw, const float* __restrict__ cb,
    const _Float16* __restrict__ btab, int useTab,
    float* __restrict__ out) {
  __shared__ _Float16 ce[EE * 16 * W_];          // [e][word][w]      32 KB
  __shared__ _Float16 cwl[KK * OUTC];            // fallback weights  1.5 KB
  __shared__ float    ybuf[NWAVE * NOUTP * 16];  // per-wave [n][word] 28 KB

  const int tid  = threadIdx.x;
  const int lane = tid & 31;
  const int wave = tid >> 5;
  const int tile = blockIdx.x;                   // 256 tiles of 16 words
  float* __restrict__ yb = &ybuf[wave * (NOUTP * 16)];

  // conv weights to LDS (fallback path only, but cheap)
  for (int i = tid; i < KK * OUTC; i += 32 * NWAVE) cwl[i] = (_Float16)cw[i];

  // gather char embeddings for the 16 words (256 table rows) into LDS,
  // transposed to [e][word][w] so A fragments are contiguous 16B DS loads.
  for (int r = tid; r < 16 * W_; r += 32 * NWAVE) {
    const int word = r >> 4, w = r & 15;
    const int g = tile * 16 + word;              // g = s*B + b
    const int s = g >> 5, b = g & 31;            // B_ == 32
    const int cidx = chars[(b * S_ + s) * W_ + w];
    const v4f* __restrict__ src = (const v4f*)(ctab + (size_t)cidx * EE);
#pragma unroll
    for (int e4 = 0; e4 < EE / 4; ++e4) {
      v4f v = src[e4];
#pragma unroll
      for (int j = 0; j < 4; ++j)
        ce[(e4 * 4 + j) * (16 * W_) + word * W_ + w] = (_Float16)v[j];
    }
  }
  __syncthreads();

  // lane roles per ISA wave32 16-bit A / B and f32 C/D layouts
  const int m     = lane & 15;            // A row (word)
  const int wb    = (lane < 16) ? 0 : 8;  // A: unused (K-half picks channel e0+keoff)
                                          // C/D: M-offset of lane's rows
  const int keoff = (lane < 16) ? 0 : 1;  // B: channel-half of lane's K rows
  const int nloc  = lane & 15;            // B/D: column within N-tile

  for (int i = 0; i < NPAIR / NWAVE; ++i) {
    const int p  = wave * (NPAIR / NWAVE) + i;
    const int e0 = 2 * p, e1 = 2 * p + 1;

    // A fragment: two contiguous 16B LDS loads
    v8h alo = *(const v8h*)&ce[(e0 + keoff) * (16 * W_) + m * W_];
    v8h ahi = *(const v8h*)&ce[(e0 + keoff) * (16 * W_) + m * W_ + 8];
    v16h a = __builtin_shufflevector(alo, ahi, 0, 1, 2, 3, 4, 5, 6, 7,
                                     8, 9, 10, 11, 12, 13, 14, 15);

#pragma unroll
    for (int tau = 0; tau < NTILE; ++tau) {
      const int ng = tau * 16 + nloc;     // this lane's output column
      v16h bv;
      if (useTab) {
        // one 32B L2-hot load, identical for all word-tiles
        bv = *(const v16h*)&btab[(((size_t)p * NTILE + tau) * 32 + lane) * 16];
      } else {
        const int e_off = (ng >= NT * FF) ? 1 : 0;
        const int r2    = ng - e_off * (NT * FF);
        const int t     = r2 >> 2;
        const int f     = r2 & 3;
        const int ch    = 2 * p + e_off;
        _Float16 w0 = cwl[0 * OUTC + ch * FF + f];
        _Float16 w1 = cwl[1 * OUTC + ch * FF + f];
        _Float16 w2 = cwl[2 * OUTC + ch * FF + f];
        if (e_off != keoff) { w0 = (_Float16)0; w1 = (_Float16)0; w2 = (_Float16)0; }
#pragma unroll
        for (int q = 0; q < 16; ++q) {
          _Float16 v = (_Float16)0;
          v = (q == t)     ? w0 : v;
          v = (q == t + 1) ? w1 : v;
          v = (q == t + 2) ? w2 : v;
          bv[q] = v;
        }
      }

      v8f c = {};
      v8f d = __builtin_amdgcn_wmma_f32_16x16x32_f16(
          false, a, false, bv, (short)0, c, false, false);

      // D: lane holds column ng for words wb..wb+7
#pragma unroll
      for (int r = 0; r < 8; ++r) yb[ng * 16 + wb + r] = d[r];
    }
    __syncthreads();   // uniform: every wave iterates NPAIR/NWAVE times

    // max over t, add bias, write this pair's 8 channels for 16 words
    {
      const int word  = lane & 15;
      const int e_off = lane >> 4;
      const int ch    = 2 * p + e_off;
      const int g     = tile * 16 + word;
      float* __restrict__ orow = out + (size_t)g * ROWF + D_;
#pragma unroll
      for (int f = 0; f < FF; ++f) {
        float mx = -__builtin_inff();
#pragma unroll
        for (int t = 0; t < NT; ++t)
          mx = fmaxf(mx, yb[(e_off * (NT * FF) + t * 4 + f) * 16 + word]);
        orow[ch * FF + f] = mx + cb[ch * FF + f];
      }
    }
    __syncthreads();   // ybuf WAR before next pair
  }
}

// ---------------------------------------------------------------------------
extern "C" void kernel_launch(void* const* d_in, const int* in_sizes, int n_in,
                              void* d_out, int out_size, void* d_ws,
                              size_t ws_size, hipStream_t stream) {
  const int*   words = (const int*)d_in[0];   // [S,B]
  const int*   chars = (const int*)d_in[1];   // [B,S,W]
  const float* wtab  = (const float*)d_in[2]; // [V,D]
  const float* ctab  = (const float*)d_in[3]; // [CV,E]
  const float* cw    = (const float*)d_in[4]; // [K,1,E*F]
  const float* cb    = (const float*)d_in[5]; // [E*F]
  float* out = (float*)d_out;                 // [S,B,556]

  const int useTab = (d_ws != nullptr && ws_size >= (size_t)BTAB_BYTES) ? 1 : 0;
  _Float16* btab = (_Float16*)d_ws;

  if (useTab) {
    const int n = NPAIR * NTILE * 32;
    build_btab_kernel<<<(n + 255) / 256, 256, 0, stream>>>(cw, btab);
  }
  word_gather_kernel<<<S_ * B_, 96, 0, stream>>>(words, wtab, out);
  char_cnn_wmma_kernel<<<(S_ * B_) / 16, 32 * NWAVE, 0, stream>>>(
      chars, ctab, cw, cb, btab, useTab, out);
}
